// SAContrastiveAdversarialLoss_12816182411399
// MI455X (gfx1250) — matrix operand order repaired and checked by reading, not compiled, not run on hardware
//
#include <hip/hip_runtime.h>
#include <hip/hip_bf16.h>
#include <math.h>

// ---- types ----
typedef __attribute__((ext_vector_type(16))) _Float16 v16h;
typedef __attribute__((ext_vector_type(8)))  _Float16 v8h;
typedef __attribute__((ext_vector_type(4)))  _Float16 v4h;
typedef __attribute__((ext_vector_type(8)))  float    v8f;

#define B_ROWS 8192
#define D_FEAT 128
#define S_LAB  64
#define TAU_INV 10.0f
#define LAM 1.0f

// LDS byte-offset of a __shared__ object (addrspacecast generic->AS3, then ptrtoint)
__device__ inline unsigned lds_off(const void* p) {
    return (unsigned)(unsigned long long)(__attribute__((address_space(3))) const void*)p;
}

// -------- async global -> LDS stage: 32 bytes per calling thread (2x B128) --------
__device__ inline void stage_async32(unsigned ldsAddr, const char* gptr) {
    asm volatile(
        "global_load_async_to_lds_b128 %0, %1, off\n\t"
        "global_load_async_to_lds_b128 %0, %1, off offset:16"
        :: "v"(ldsAddr), "v"(gptr)
        : "memory");
}

// -------- 8x plain LDS b128 loads (8 halves each, 32-byte steps) from one per-lane base address ------
__device__ inline void ds_tr16_x8(unsigned addr,
                                  v8h& f0, v8h& f1, v8h& f2, v8h& f3,
                                  v8h& f4, v8h& f5, v8h& f6, v8h& f7) {
    asm volatile(
        "ds_load_b128 %0, %8 offset:0\n\t"
        "ds_load_b128 %1, %8 offset:32\n\t"
        "ds_load_b128 %2, %8 offset:64\n\t"
        "ds_load_b128 %3, %8 offset:96\n\t"
        "ds_load_b128 %4, %8 offset:128\n\t"
        "ds_load_b128 %5, %8 offset:160\n\t"
        "ds_load_b128 %6, %8 offset:192\n\t"
        "ds_load_b128 %7, %8 offset:224\n\t"
        "s_wait_dscnt 0x0"
        : "=&v"(f0), "=&v"(f1), "=&v"(f2), "=&v"(f3),
          "=&v"(f4), "=&v"(f5), "=&v"(f6), "=&v"(f7)
        : "v"(addr)
        : "memory");
}

// -------- zero accumulators + output --------
__global__ void sacl_init_kernel(float* acc, int n, float* out) {
    int i = blockIdx.x * blockDim.x + threadIdx.x;
    if (i < n) acc[i] = 0.0f;
    if (i == 0) out[0] = 0.0f;
}

// -------- row L2-normalize fp32 -> f16 (wave per row, 128 feats) --------
__global__ __launch_bounds__(256)
void sacl_normalize_kernel(const float* __restrict__ x, _Float16* __restrict__ y) {
    const int lane = threadIdx.x & 31;
    const int wave = threadIdx.x >> 5;
    const int row  = blockIdx.x * 8 + wave;
    const float4* px = (const float4*)(x + (size_t)row * D_FEAT);
    float4 v = px[lane];
    float ss = v.x * v.x + v.y * v.y + v.z * v.z + v.w * v.w;
    #pragma unroll
    for (int m = 16; m >= 1; m >>= 1) ss += __shfl_xor(ss, m, 32);
    float inv = 1.0f / fmaxf(sqrtf(ss), 1e-8f);
    v4h h;
    h.x = (_Float16)(v.x * inv);
    h.y = (_Float16)(v.y * inv);
    h.z = (_Float16)(v.z * inv);
    h.w = (_Float16)(v.w * inv);
    *(v4h*)(y + (size_t)row * D_FEAT + lane * 4) = h;
}

// -------- fused S12/S11: LDS double-buffered, async-staged, WMMA + exp + row-sum -------
// Block: 256 threads = 8 waves; block covers 128 rows (wave w -> rows blockIdx.x*128+w*16..).
// Column tiles staged once per block into LDS and shared by all 8 waves (8x L2 reuse).
__global__ __launch_bounds__(256)
void sacl_fused_kernel(const _Float16* __restrict__ n1h,
                       const _Float16* __restrict__ n2h,
                       float* __restrict__ rowsum12,
                       float* __restrict__ rowsum11,
                       float* __restrict__ numer) {
    // [buf][mat: 0=n2, 1=n1][16 rows * 128 halves] = 16 KB
    __shared__ _Float16 lds[2][2][16 * D_FEAT];

    const int tid  = threadIdx.x;
    const int lane = tid & 31;
    const int wave = tid >> 5;
    const int r0   = lane & 15;
    const int g    = lane >> 4;
    const int rowBase = blockIdx.x * 128 + wave * 16;
    const int jBegin  = blockIdx.y * 32;   // 16 chunks x 32 col-tiles = 512 tiles

    const unsigned ldsBase = lds_off(&lds[0][0][0]);

    // ---- staging role: mat = tid>>7 (0:n2, 1:n1), 32 bytes per thread per tile ----
    const int smat = tid >> 7;
    const int sidx = tid & 127;
    const char* gsrc0 = (const char*)(smat ? n1h : n2h) +
                        (size_t)jBegin * 4096 + sidx * 32;
    const unsigned sdst0 = ldsBase + smat * 4096 + sidx * 32;   // + buf*8192

    // ---- A fragments: 16x32 f16 layout ----
    const _Float16* Arow = n1h + (size_t)(rowBase + r0) * D_FEAT;
    v16h afrag[4];
    #pragma unroll
    for (int kb = 0; kb < 4; ++kb) {
        v8h lo = *(const v8h*)(Arow + kb * 32 + g * 8);
        v8h hi = *(const v8h*)(Arow + kb * 32 + 16 + g * 8);
        afrag[kb] = __builtin_shufflevector(lo, hi,
            0, 1, 2, 3, 4, 5, 6, 7, 8, 9, 10, 11, 12, 13, 14, 15);
    }

    float s12[8], s11[8], nm[8];
    #pragma unroll
    for (int r = 0; r < 8; ++r) { s12[r] = 0.f; s11[r] = 0.f; nm[r] = 0.f; }

    // per-lane LDS base inside a staged 16x128 tile (immediate offsets select kb/half)
    const unsigned tileLane = (unsigned)(r0 * 256 + g * 16);

    // ---- prologue: stage tile 0 into buffer 0 ----
    stage_async32(sdst0, gsrc0);
    asm volatile("s_wait_asynccnt 0x0" ::: "memory");
    __syncthreads();

    for (int jt = 0; jt < 32; ++jt) {
        const int buf = jt & 1;
        // prefetch next tile into the other buffer (overlaps with WMMA below)
        if (jt + 1 < 32) {
            stage_async32(sdst0 + (buf ^ 1) * 8192,
                          gsrc0 + (size_t)(jt + 1) * 4096);
        }

        const int colBase = (jBegin + jt) * 16;
        const unsigned bufBase = ldsBase + buf * 8192;

        // ---- S12: B frags from staged n2 tile ----
        v8f acc12 = {};
        {
            v8h f0, f1, f2, f3, f4, f5, f6, f7;
            ds_tr16_x8(bufBase + tileLane, f0, f1, f2, f3, f4, f5, f6, f7);
            v16h b0 = __builtin_shufflevector(f0, f1, 0,1,2,3,4,5,6,7,8,9,10,11,12,13,14,15);
            v16h b1 = __builtin_shufflevector(f2, f3, 0,1,2,3,4,5,6,7,8,9,10,11,12,13,14,15);
            v16h b2 = __builtin_shufflevector(f4, f5, 0,1,2,3,4,5,6,7,8,9,10,11,12,13,14,15);
            v16h b3 = __builtin_shufflevector(f6, f7, 0,1,2,3,4,5,6,7,8,9,10,11,12,13,14,15);
            acc12 = __builtin_amdgcn_wmma_f32_16x16x32_f16(false, afrag[0], false, b0, (short)0, acc12, false, false);
            acc12 = __builtin_amdgcn_wmma_f32_16x16x32_f16(false, afrag[1], false, b1, (short)0, acc12, false, false);
            acc12 = __builtin_amdgcn_wmma_f32_16x16x32_f16(false, afrag[2], false, b2, (short)0, acc12, false, false);
            acc12 = __builtin_amdgcn_wmma_f32_16x16x32_f16(false, afrag[3], false, b3, (short)0, acc12, false, false);
        }

        // ---- S11: B frags from staged n1 tile ----
        v8f acc11 = {};
        {
            v8h f0, f1, f2, f3, f4, f5, f6, f7;
            ds_tr16_x8(bufBase + 4096 + tileLane, f0, f1, f2, f3, f4, f5, f6, f7);
            v16h b0 = __builtin_shufflevector(f0, f1, 0,1,2,3,4,5,6,7,8,9,10,11,12,13,14,15);
            v16h b1 = __builtin_shufflevector(f2, f3, 0,1,2,3,4,5,6,7,8,9,10,11,12,13,14,15);
            v16h b2 = __builtin_shufflevector(f4, f5, 0,1,2,3,4,5,6,7,8,9,10,11,12,13,14,15);
            v16h b3 = __builtin_shufflevector(f6, f7, 0,1,2,3,4,5,6,7,8,9,10,11,12,13,14,15);
            acc11 = __builtin_amdgcn_wmma_f32_16x16x32_f16(false, afrag[0], false, b0, (short)0, acc11, false, false);
            acc11 = __builtin_amdgcn_wmma_f32_16x16x32_f16(false, afrag[1], false, b1, (short)0, acc11, false, false);
            acc11 = __builtin_amdgcn_wmma_f32_16x16x32_f16(false, afrag[2], false, b2, (short)0, acc11, false, false);
            acc11 = __builtin_amdgcn_wmma_f32_16x16x32_f16(false, afrag[3], false, b3, (short)0, acc11, false, false);
        }

        // ---- epilogue: 1/tau scale, exp, diag handling; C/D layout: M=r+8g, N=r0 ----
        const int Ng = colBase + r0;
        #pragma unroll
        for (int r = 0; r < 8; ++r) {
            const int Mg = rowBase + r + 8 * g;
            float e12 = __expf(acc12[r] * TAU_INV);
            float e11 = __expf(acc11[r] * TAU_INV);
            bool diag = (Mg == Ng);
            s12[r] += e12;
            s11[r] += diag ? 0.0f : e11;
            nm[r]  += diag ? e12 : 0.0f;
        }

        // make prefetched buffer visible to all waves before next iteration
        asm volatile("s_wait_asynccnt 0x0" ::: "memory");
        __syncthreads();
    }

    // ---- reduce over N within each 16-lane half, one atomic per row ----
    #pragma unroll
    for (int r = 0; r < 8; ++r) {
        float v12 = s12[r], v11 = s11[r], vn = nm[r];
        #pragma unroll
        for (int m = 8; m >= 1; m >>= 1) {
            v12 += __shfl_xor(v12, m, 16);
            v11 += __shfl_xor(v11, m, 16);
            vn  += __shfl_xor(vn,  m, 16);
        }
        if (r0 == 0) {
            const int M = rowBase + r + 8 * g;
            atomicAdd(&rowsum12[M], v12);
            atomicAdd(&rowsum11[M], v11);
            atomicAdd(&numer[M],    vn);
        }
    }
}

// -------- contrastive finalize --------
__global__ __launch_bounds__(256)
void sacl_finalize_kernel(const float* __restrict__ rs12,
                          const float* __restrict__ rs11,
                          const float* __restrict__ nm,
                          float* __restrict__ out) {
    int i = blockIdx.x * blockDim.x + threadIdx.x;
    if (i >= B_ROWS) return;
    float denom = rs12[i] + rs11[i];
    float c = -logf(1e-12f + nm[i] / denom);
    c *= 1.0f / ((float)B_ROWS * (2.0f * (float)B_ROWS - 1.0f));
    atomicAdd(out, c);
}

// -------- adversarial term: wave per row over 64 labels --------
__global__ __launch_bounds__(256)
void sacl_adversarial_kernel(const float* __restrict__ c,
                             const float* __restrict__ lab,
                             float* __restrict__ out) {
    const int lane = threadIdx.x & 31;
    const int wave = threadIdx.x >> 5;
    const int row  = blockIdx.x * 8 + wave;
    const float* cr = c + (size_t)row * S_LAB;
    const float* lr = lab + (size_t)row * S_LAB;
    float c0 = cr[lane], c1 = cr[lane + 32];
    float l0 = lr[lane], l1 = lr[lane + 32];
    float ss = c0 * c0 + c1 * c1;
    float bv; int bi;
    if (l0 >= l1) { bv = l0; bi = lane; } else { bv = l1; bi = lane + 32; }
    #pragma unroll
    for (int m = 16; m >= 1; m >>= 1) {
        ss += __shfl_xor(ss, m, 32);
        float ov = __shfl_xor(bv, m, 32);
        int   oi = __shfl_xor(bi, m, 32);
        if (ov > bv || (ov == bv && oi < bi)) { bv = ov; bi = oi; }
    }
    if (lane == 0) {
        float picked = cr[bi] / fmaxf(sqrtf(ss), 1e-12f);
        atomicAdd(out, LAM * (-logf(1e-12f + 1.0f - picked)));
    }
}

extern "C" void kernel_launch(void* const* d_in, const int* in_sizes, int n_in,
                              void* d_out, int out_size, void* d_ws, size_t ws_size,
                              hipStream_t stream) {
    const float* z1  = (const float*)d_in[0];
    const float* z2  = (const float*)d_in[1];
    const float* c   = (const float*)d_in[2];
    const float* lab = (const float*)d_in[3];
    float* out = (float*)d_out;

    char* ws = (char*)d_ws;
    _Float16* n1h = (_Float16*)ws;                              // 2 MB
    _Float16* n2h = (_Float16*)(ws + (size_t)2 * 1024 * 1024);  // 2 MB
    float* rs12 = (float*)(ws + (size_t)4 * 1024 * 1024);
    float* rs11 = rs12 + B_ROWS;
    float* nm   = rs11 + B_ROWS;

    sacl_init_kernel<<<(3 * B_ROWS + 255) / 256, 256, 0, stream>>>(rs12, 3 * B_ROWS, out);
    sacl_normalize_kernel<<<B_ROWS / 8, 256, 0, stream>>>(z1, n1h);
    sacl_normalize_kernel<<<B_ROWS / 8, 256, 0, stream>>>(z2, n2h);
    dim3 grid(B_ROWS / 128, 16);
    sacl_fused_kernel<<<grid, 256, 0, stream>>>(n1h, n2h, rs12, rs11, nm);
    sacl_finalize_kernel<<<B_ROWS / 256, 256, 0, stream>>>(rs12, rs11, nm, out);
    sacl_adversarial_kernel<<<B_ROWS / 8, 256, 0, stream>>>(c, lab, out);
}
